// TestSiluMulFp8QuantModel_2886218023600
// MI455X (gfx1250) — 8-bit matrix operand order repaired and checked by reading, compiled, not run on hardware
//
#include <hip/hip_runtime.h>
#include <hip/hip_bf16.h>

typedef __attribute__((ext_vector_type(16))) int   v16i;
typedef __attribute__((ext_vector_type(8)))  float v8f;

#define BM 128
#define BN 128
#define BK 128

__global__ __launch_bounds__(256) void silu_quant_kernel(
    const float* __restrict__ x, const float* __restrict__ wscale,
    unsigned char* __restrict__ q, int T, int H)
{
    long long idx = ((long long)blockIdx.x * blockDim.x + threadIdx.x) * 4;
    long long total = (long long)T * H;
    if (idx >= total) return;
    int t = (int)(idx / H);
    int h = (int)(idx % H);

    const float4 g = *(const float4*)(x + (long long)t * 2 * H + h);
    const float4 u = *(const float4*)(x + (long long)t * 2 * H + H + h);
    float inv_s = 1.0f / wscale[0];

    float y0 = (g.x / (1.0f + __expf(-g.x))) * u.x * inv_s;
    float y1 = (g.y / (1.0f + __expf(-g.y))) * u.y * inv_s;
    float y2 = (g.z / (1.0f + __expf(-g.z))) * u.z * inv_s;
    float y3 = (g.w / (1.0f + __expf(-g.w))) * u.w * inv_s;

    y0 = fminf(fmaxf(y0, -448.0f), 448.0f);
    y1 = fminf(fmaxf(y1, -448.0f), 448.0f);
    y2 = fminf(fmaxf(y2, -448.0f), 448.0f);
    y3 = fminf(fmaxf(y3, -448.0f), 448.0f);

    int p = __builtin_amdgcn_cvt_pk_fp8_f32(y0, y1, 0, false);
    p     = __builtin_amdgcn_cvt_pk_fp8_f32(y2, y3, p, true);
    *(unsigned int*)(q + idx) = (unsigned int)p;
}

__global__ __launch_bounds__(256) void w_to_fp8T_kernel(
    const float* __restrict__ w, unsigned char* __restrict__ wT, int H)
{
    __shared__ unsigned char tile[64][68];
    int kt = blockIdx.y * 64;
    int nt = blockIdx.x * 64;

#pragma unroll
    for (int r = 0; r < 4; ++r) {
        int lin = threadIdx.x + r * 256;
        int k   = lin >> 4;
        int n4  = (lin & 15) << 2;
        float4 v = *(const float4*)(w + (long long)(kt + k) * H + nt + n4);
        int p = __builtin_amdgcn_cvt_pk_fp8_f32(v.x, v.y, 0, false);
        p     = __builtin_amdgcn_cvt_pk_fp8_f32(v.z, v.w, p, true);
        tile[n4 + 0][k] = (unsigned char)( p        & 0xff);
        tile[n4 + 1][k] = (unsigned char)((p >> 8)  & 0xff);
        tile[n4 + 2][k] = (unsigned char)((p >> 16) & 0xff);
        tile[n4 + 3][k] = (unsigned char)((p >> 24) & 0xff);
    }
    __syncthreads();
#pragma unroll
    for (int r = 0; r < 4; ++r) {
        int lin = threadIdx.x + r * 256;
        int n   = lin >> 4;
        int kd  = (lin & 15) << 2;
        unsigned int d = *(const unsigned int*)&tile[n][kd];
        *(unsigned int*)(wT + (long long)(nt + n) * H + kt + kd) = d;
    }
}

__global__ __launch_bounds__(256) void gemm_fp8_kernel(
    const unsigned char* __restrict__ q,
    const unsigned char* __restrict__ wT,
    const float* __restrict__ wscale,
    float* __restrict__ out,
    int M, int N, int K)
{
    __shared__ alignas(16) unsigned char As[2][BM * BK];
    __shared__ alignas(16) unsigned char Bs[2][BN * BK];

    const int m0   = blockIdx.y * BM;
    const int n0   = blockIdx.x * BN;
    const int wave = threadIdx.x >> 5;
    const int lane = threadIdx.x & 31;
    const int hl   = lane >> 4;
    const int l16  = lane & 15;

    v8f acc[8] = {};

    auto stage_async = [&](int kb, int nbuf) {
#pragma unroll
        for (int r = 0; r < 4; ++r) {
            int lin = threadIdx.x + r * 256;
            int row = lin >> 3;
            int kk  = (lin & 7) << 4;
            unsigned lA = (unsigned)(size_t)&As[nbuf][row * BK + kk];
            unsigned lB = (unsigned)(size_t)&Bs[nbuf][row * BK + kk];
            const unsigned char* gA = q  + (long long)(m0 + row) * K + kb + kk;
            const unsigned char* gB = wT + (long long)(n0 + row) * K + kb + kk;
            asm volatile("global_load_async_to_lds_b128 %0, %1, off"
                         :: "v"(lA), "v"(gA) : "memory");
            asm volatile("global_load_async_to_lds_b128 %0, %1, off"
                         :: "v"(lB), "v"(gB) : "memory");
        }
    };
    auto wait_async = [&]() {
        asm volatile("s_wait_asynccnt 0x0" ::: "memory");
    };

    auto loadA = [&](int buf) {
        const unsigned char* ab = &As[buf][(wave * 16 + l16) * BK];
        v16i a;
#pragma unroll
        for (int s = 0; s < 2; ++s) {
            uint2 p0 = *(const uint2*)(ab + s * 64 + hl * 8 + 0);
            uint2 p1 = *(const uint2*)(ab + s * 64 + hl * 8 + 16);
            uint2 p2 = *(const uint2*)(ab + s * 64 + hl * 8 + 32);
            uint2 p3 = *(const uint2*)(ab + s * 64 + hl * 8 + 48);
            a[s * 8 + 0] = (int)p0.x;  a[s * 8 + 1] = (int)p0.y;
            a[s * 8 + 2] = (int)p1.x;  a[s * 8 + 3] = (int)p1.y;
            a[s * 8 + 4] = (int)p2.x;  a[s * 8 + 5] = (int)p2.y;
            a[s * 8 + 6] = (int)p3.x;  a[s * 8 + 7] = (int)p3.y;
        }
        return a;
    };
    auto loadB = [&](int buf, int t) {
        const unsigned char* bb = &Bs[buf][(t * 16 + l16) * BK];
        v16i b;
#pragma unroll
        for (int g = 0; g < 4; ++g) {
            uint2 v0 = *(const uint2*)(bb + g * 32 + hl * 8);
            uint2 v1 = *(const uint2*)(bb + g * 32 + hl * 8 + 16);
            b[g * 4 + 0] = (int)v0.x;  b[g * 4 + 1] = (int)v0.y;
            b[g * 4 + 2] = (int)v1.x;  b[g * 4 + 3] = (int)v1.y;
        }
        return b;
    };

    const int nSteps = K / BK;
    stage_async(0, 0);
    wait_async();
    __syncthreads();

    for (int ks = 0; ks < nSteps; ++ks) {
        const int buf = ks & 1;
        if (ks + 1 < nSteps) stage_async((ks + 1) * BK, buf ^ 1);

        v16i a = loadA(buf);

        v16i b0 = loadB(buf, 0);
        v16i b1 = loadB(buf, 1);
        v16i b2 = loadB(buf, 2);
        v16i b3 = loadB(buf, 3);
        __builtin_amdgcn_sched_barrier(0);
        acc[0] = __builtin_amdgcn_wmma_f32_16x16x128_fp8_fp8(a, b0, (short)0, acc[0], false, false);
        acc[1] = __builtin_amdgcn_wmma_f32_16x16x128_fp8_fp8(a, b1, (short)0, acc[1], false, false);
        acc[2] = __builtin_amdgcn_wmma_f32_16x16x128_fp8_fp8(a, b2, (short)0, acc[2], false, false);
        acc[3] = __builtin_amdgcn_wmma_f32_16x16x128_fp8_fp8(a, b3, (short)0, acc[3], false, false);

        b0 = loadB(buf, 4);
        b1 = loadB(buf, 5);
        b2 = loadB(buf, 6);
        b3 = loadB(buf, 7);
        __builtin_amdgcn_sched_barrier(0);
        acc[4] = __builtin_amdgcn_wmma_f32_16x16x128_fp8_fp8(a, b0, (short)0, acc[4], false, false);
        acc[5] = __builtin_amdgcn_wmma_f32_16x16x128_fp8_fp8(a, b1, (short)0, acc[5], false, false);
        acc[6] = __builtin_amdgcn_wmma_f32_16x16x128_fp8_fp8(a, b2, (short)0, acc[6], false, false);
        acc[7] = __builtin_amdgcn_wmma_f32_16x16x128_fp8_fp8(a, b3, (short)0, acc[7], false, false);

        if (ks + 1 < nSteps) wait_async();
        __syncthreads();
    }

    const float s  = wscale[0];
    const float s2 = s * s;
#pragma unroll
    for (int t = 0; t < 8; ++t) {
#pragma unroll
        for (int r = 0; r < 8; ++r) {
            int gm = m0 + wave * 16 + hl * 8 + r;
            int gn = n0 + t * 16 + l16;
            out[(long long)gm * N + gn] = acc[t][r] * s2;
        }
    }
}

extern "C" void kernel_launch(void* const* d_in, const int* in_sizes, int n_in,
                              void* d_out, int out_size, void* d_ws, size_t ws_size,
                              hipStream_t stream) {
    const float* x      = (const float*)d_in[0];
    const float* w      = (const float*)d_in[1];
    const float* wscale = (const float*)d_in[2];
    float* out          = (float*)d_out;
    (void)n_in; (void)out_size; (void)ws_size;

    int H = 1;
    while ((long long)H * H < (long long)in_sizes[1]) H <<= 1;
    int T = (int)(in_sizes[0] / (long long)(2 * H));

    unsigned char* q  = (unsigned char*)d_ws;
    unsigned char* wT = q + (size_t)T * H;

    long long nPack = (long long)T * H / 4;
    silu_quant_kernel<<<dim3((unsigned)((nPack + 255) / 256)), dim3(256), 0, stream>>>(
        x, wscale, q, T, H);

    w_to_fp8T_kernel<<<dim3(H / 64, H / 64), dim3(256), 0, stream>>>(w, wT, H);

    gemm_fp8_kernel<<<dim3(H / BN, T / BM), dim3(256), 0, stream>>>(
        q, wT, wscale, out, T, H, H);
}
